// StepwiseSudokuGPT_27994596836246
// MI455X (gfx1250) — matrix operand order repaired and checked by reading, not compiled, not run on hardware
//
#include <hip/hip_runtime.h>
#include <hip/hip_bf16.h>
#include <cstdint>
#include <cstddef>

// ---------------------------------------------------------------------------
// StepwiseSudokuGPT on MI455X (gfx1250): bf16 WMMA GEMMs, fp32 accumulate.
// B=32, L=12, H=12, C=768, HD=64, NC=81, passes T = 82..86.
// GEMM K-loop is software-pipelined: fragments for step k+32 are loaded while
// the 4 WMMAs for step k execute, so s_wait_loadcnt overlaps the matrix pipe.
// ---------------------------------------------------------------------------

typedef __bf16 bf16_t;
typedef __attribute__((ext_vector_type(16))) __bf16 v16bf;
typedef __attribute__((ext_vector_type(8)))  __bf16 v8bf;
typedef __attribute__((ext_vector_type(8)))  float  v8f;

__device__ __forceinline__ v16bf load_frag_a(const bf16_t* __restrict__ row,
                                             int k0, int half) {
  // A fragment in the contiguous k order used here: lane holds row m; element e -> k = half*16 + e
  union { v16bf v; v8bf h[2]; } u;
  u.h[0] = *reinterpret_cast<const v8bf*>(row + k0 + half * 16);
  u.h[1] = *reinterpret_cast<const v8bf*>(row + k0 + half * 16 + 8);
  return u.v;
}

__device__ __forceinline__ v16bf load_frag_b(const bf16_t* __restrict__ row,
                                             int k0, int half) {
  // B fragment in the same contiguous k order as A: lane holds col n; element e -> k = half*16 + e.
  // Bt stored transposed (N x K) so this is 16 contiguous bf16 (2 x b128).
  union { v16bf v; v8bf h[2]; } u;
  const bf16_t* p = row + k0 + half * 16;
  u.h[0] = reinterpret_cast<const v8bf*>(p)[0];
  u.h[1] = reinterpret_cast<const v8bf*>(p)[1];
  return u.v;
}

// MODE 0: out_f32 = A@B + bias            (qkv)
// MODE 1: out_f32 = resid + A@B + bias    (proj / mlp2 residual add, in-place ok)
// MODE 2: out_bf16 = gelu(A@B + bias)     (mlp1)
template <int MODE>
__global__ __launch_bounds__(256) void gemm_bf16_wmma(
    const bf16_t* __restrict__ A, const bf16_t* __restrict__ Bt,
    const float* __restrict__ bias, const float* __restrict__ resid,
    float* __restrict__ outF, bf16_t* __restrict__ outB,
    int M, int K, int N) {
  const int lane = threadIdx.x & 31;
  const int wave = threadIdx.x >> 5;
  const int half = lane >> 4;
  const int ln16 = lane & 15;
  const int m0 = blockIdx.x * 128 + wave * 16;   // wave-uniform
  const int n0 = blockIdx.y * 64;
  if (m0 >= M) return;                           // wave-uniform exit; EXEC stays full

  const bf16_t* __restrict__ Arow = A + (size_t)(m0 + ln16) * K;
  const bf16_t* __restrict__ Brow0 = Bt + (size_t)(n0 + 0 * 16 + ln16) * K;
  const bf16_t* __restrict__ Brow1 = Bt + (size_t)(n0 + 1 * 16 + ln16) * K;
  const bf16_t* __restrict__ Brow2 = Bt + (size_t)(n0 + 2 * 16 + ln16) * K;
  const bf16_t* __restrict__ Brow3 = Bt + (size_t)(n0 + 3 * 16 + ln16) * K;

  v8f acc[4] = {};

  // ---- software-pipelined K loop (steps of 32) ----
  v16bf a_cur = load_frag_a(Arow, 0, half);
  v16bf b_cur0 = load_frag_b(Brow0, 0, half);
  v16bf b_cur1 = load_frag_b(Brow1, 0, half);
  v16bf b_cur2 = load_frag_b(Brow2, 0, half);
  v16bf b_cur3 = load_frag_b(Brow3, 0, half);

  for (int k0 = 32; k0 < K; k0 += 32) {
    // issue next-step fragment loads first (no dependence on current WMMAs)
    __builtin_prefetch(Arow + k0 + 32, 0, 3);
    __builtin_prefetch(Brow0 + k0 + 32, 0, 3);
    v16bf a_nxt  = load_frag_a(Arow, k0, half);
    v16bf b_nxt0 = load_frag_b(Brow0, k0, half);
    v16bf b_nxt1 = load_frag_b(Brow1, k0, half);
    v16bf b_nxt2 = load_frag_b(Brow2, k0, half);
    v16bf b_nxt3 = load_frag_b(Brow3, k0, half);

    acc[0] = __builtin_amdgcn_wmma_f32_16x16x32_bf16(false, a_cur, false, b_cur0,
                                                     (short)0, acc[0], false, false);
    acc[1] = __builtin_amdgcn_wmma_f32_16x16x32_bf16(false, a_cur, false, b_cur1,
                                                     (short)0, acc[1], false, false);
    acc[2] = __builtin_amdgcn_wmma_f32_16x16x32_bf16(false, a_cur, false, b_cur2,
                                                     (short)0, acc[2], false, false);
    acc[3] = __builtin_amdgcn_wmma_f32_16x16x32_bf16(false, a_cur, false, b_cur3,
                                                     (short)0, acc[3], false, false);

    a_cur = a_nxt;
    b_cur0 = b_nxt0; b_cur1 = b_nxt1; b_cur2 = b_nxt2; b_cur3 = b_nxt3;
  }
  // epilogue step
  acc[0] = __builtin_amdgcn_wmma_f32_16x16x32_bf16(false, a_cur, false, b_cur0,
                                                   (short)0, acc[0], false, false);
  acc[1] = __builtin_amdgcn_wmma_f32_16x16x32_bf16(false, a_cur, false, b_cur1,
                                                   (short)0, acc[1], false, false);
  acc[2] = __builtin_amdgcn_wmma_f32_16x16x32_bf16(false, a_cur, false, b_cur2,
                                                   (short)0, acc[2], false, false);
  acc[3] = __builtin_amdgcn_wmma_f32_16x16x32_bf16(false, a_cur, false, b_cur3,
                                                   (short)0, acc[3], false, false);

#pragma unroll
  for (int j = 0; j < 4; ++j) {
    const int n = n0 + j * 16 + ln16;
    const float bs = bias[n];
#pragma unroll
    for (int r = 0; r < 8; ++r) {
      const int m = m0 + r + half * 8;           // C/D layout: vgpr r, lane-half
      const size_t o = (size_t)m * N + n;
      float v = acc[j][r] + bs;
      if (MODE == 0) {
        outF[o] = v;
      } else if (MODE == 1) {
        outF[o] = resid[o] + v;
      } else {
        float g = 0.5f * v * (1.0f + erff(v * 0.70710678118654752f));
        outB[o] = (bf16_t)g;
      }
    }
  }
}

// fp32 (K x N) -> bf16 transposed (N x K), L matrices stacked.
__global__ void transpose_to_bf16(const float* __restrict__ W, bf16_t* __restrict__ Wt,
                                  int K, int N, int Lc) {
  size_t i = (size_t)blockIdx.x * blockDim.x + threadIdx.x;
  size_t per = (size_t)K * N;
  size_t total = per * Lc;
  if (i >= total) return;
  size_t l = i / per;
  size_t rem = i - l * per;
  int k = (int)(rem / N);
  int n = (int)(rem - (size_t)k * N);
  Wt[l * per + (size_t)n * K + k] = (bf16_t)W[i];
}

// Build input embeddings for pass p (p cot tokens already produced).
__global__ void embed_kernel(const int* __restrict__ idx, const float* __restrict__ wte,
                             const float* __restrict__ special, const float* __restrict__ cot,
                             int T, float* __restrict__ x) {
  int i = blockIdx.x * blockDim.x + threadIdx.x;
  int total = 32 * T * 768;
  if (i >= total) return;
  int c = i % 768;
  int bt = i / 768;
  int t = bt % T;
  int b = bt / T;
  float v;
  if (t < 81) {
    v = wte[(size_t)idx[b * 81 + t] * 768 + c];
  } else if (t < T - 1) {
    v = cot[((size_t)(t - 81) * 32 + b) * 768 + c];
  } else {
    v = special[c];
  }
  x[i] = v;
}

// LayerNorm of one row (block per row), write bf16 activations.
__global__ __launch_bounds__(256) void layernorm_bf16(
    const float* __restrict__ x, const float* __restrict__ g,
    const float* __restrict__ bta, bf16_t* __restrict__ h) {
  const int row = blockIdx.x;
  const int tid = threadIdx.x;
  const float* xr = x + (size_t)row * 768;
  __shared__ float s1[256], s2[256];
  float a = 0.f, b = 0.f;
  for (int c = tid; c < 768; c += 256) { float v = xr[c]; a += v; b += v * v; }
  s1[tid] = a; s2[tid] = b;
  __syncthreads();
  for (int off = 128; off > 0; off >>= 1) {
    if (tid < off) { s1[tid] += s1[tid + off]; s2[tid] += s2[tid + off]; }
    __syncthreads();
  }
  const float mean = s1[0] * (1.0f / 768.0f);
  const float var = s2[0] * (1.0f / 768.0f) - mean * mean;
  const float r = rsqrtf(var + 1e-5f);
  for (int c = tid; c < 768; c += 256)
    h[(size_t)row * 768 + c] = (bf16_t)((xr[c] - mean) * r * g[c] + bta[c]);
}

// Final LN of the last token only -> fp32 (cot slot or d_out).
__global__ __launch_bounds__(256) void lnf_last(
    const float* __restrict__ x, const float* __restrict__ g,
    const float* __restrict__ bta, int T, float* __restrict__ dst) {
  const int b = blockIdx.x;
  const int tid = threadIdx.x;
  const float* xr = x + ((size_t)b * T + (T - 1)) * 768;
  __shared__ float s1[256], s2[256];
  float a = 0.f, c2 = 0.f;
  for (int c = tid; c < 768; c += 256) { float v = xr[c]; a += v; c2 += v * v; }
  s1[tid] = a; s2[tid] = c2;
  __syncthreads();
  for (int off = 128; off > 0; off >>= 1) {
    if (tid < off) { s1[tid] += s1[tid + off]; s2[tid] += s2[tid + off]; }
    __syncthreads();
  }
  const float mean = s1[0] * (1.0f / 768.0f);
  const float var = s2[0] * (1.0f / 768.0f) - mean * mean;
  const float r = rsqrtf(var + 1e-5f);
  for (int c = tid; c < 768; c += 256)
    dst[(size_t)b * 768 + c] = (xr[c] - mean) * r * g[c] + bta[c];
}

// Causal attention for one (batch, head): rope(q,k), softmax(q.k/8), @v -> bf16.
__global__ __launch_bounds__(128) void attention_kernel(
    const float* __restrict__ qkv, bf16_t* __restrict__ o, int T) {
  constexpr int TMAX = 86;
  const int b = blockIdx.x / 12;
  const int hd = blockIdx.x % 12;
  const int tid = threadIdx.x;
  __shared__ bf16_t Ks[TMAX * 64];
  __shared__ bf16_t Vs[TMAX * 64];
  __shared__ float Sc[TMAX * 88];

  const size_t base = (size_t)b * T * 2304;
  const float LOG1E4_OVER32 = 0.28782313662425572f;  // ln(10000)/32

  for (int i = tid; i < T * 64; i += 128) {
    int t = i >> 6, d = i & 63;
    const float* kr = qkv + base + (size_t)t * 2304 + 768 + hd * 64;
    int i2 = d & 31;
    float inv = __expf(-LOG1E4_OVER32 * (float)i2);
    float sn, cs;
    __sincosf((float)t * inv, &sn, &cs);
    float kv = (d < 32) ? (kr[d] * cs - kr[d + 32] * sn)
                        : (kr[d - 32] * sn + kr[d] * cs);
    Ks[i] = (bf16_t)kv;
    Vs[i] = (bf16_t)qkv[base + (size_t)t * 2304 + 1536 + hd * 64 + d];
  }
  __syncthreads();

  const int q = tid;
  if (q < T) {
    float qv[64];
    const float* qr = qkv + base + (size_t)q * 2304 + hd * 64;
#pragma unroll
    for (int d = 0; d < 64; ++d) {
      int i2 = d & 31;
      float inv = __expf(-LOG1E4_OVER32 * (float)i2);
      float sn, cs;
      __sincosf((float)q * inv, &sn, &cs);
      qv[d] = (d < 32) ? (qr[d] * cs - qr[d + 32] * sn)
                       : (qr[d - 32] * sn + qr[d] * cs);
    }
    float mx = -3.0e38f;
    for (int kk = 0; kk <= q; ++kk) {
      float s = 0.f;
#pragma unroll 8
      for (int d = 0; d < 64; ++d) s += qv[d] * (float)Ks[kk * 64 + d];
      s *= 0.125f;
      Sc[q * 88 + kk] = s;
      mx = fmaxf(mx, s);
    }
    float den = 0.f;
    for (int kk = 0; kk <= q; ++kk) {
      float e = __expf(Sc[q * 88 + kk] - mx);
      Sc[q * 88 + kk] = e;
      den += e;
    }
    const float rden = 1.0f / den;
    bf16_t* orow = o + ((size_t)(b * T + q)) * 768 + hd * 64;
#pragma unroll 4
    for (int d = 0; d < 64; ++d) {
      float a = 0.f;
      for (int kk = 0; kk <= q; ++kk) a += Sc[q * 88 + kk] * (float)Vs[kk * 64 + d];
      orow[d] = (bf16_t)(a * rden);
    }
  }
}

// ---------------------------------------------------------------------------

extern "C" void kernel_launch(void* const* d_in, const int* in_sizes, int n_in,
                              void* d_out, int out_size, void* d_ws, size_t ws_size,
                              hipStream_t stream) {
  (void)in_sizes; (void)n_in; (void)out_size; (void)ws_size;
  constexpr int Lc = 12, Cc = 768, Bc = 32, TMAX = 86, NCOT = 4; // num_cot_tokens traced as 4

  const int*   idx      = (const int*)  d_in[0];
  const float* wte      = (const float*)d_in[2];
  const float* special  = (const float*)d_in[3];
  const float* ln1_g    = (const float*)d_in[4];
  const float* ln1_b    = (const float*)d_in[5];
  const float* attn_w   = (const float*)d_in[6];
  const float* attn_b   = (const float*)d_in[7];
  const float* proj_w   = (const float*)d_in[8];
  const float* proj_b   = (const float*)d_in[9];
  const float* ln2_g    = (const float*)d_in[10];
  const float* ln2_b    = (const float*)d_in[11];
  const float* mlp_w1   = (const float*)d_in[12];
  const float* mlp_b1   = (const float*)d_in[13];
  const float* mlp_w2   = (const float*)d_in[14];
  const float* mlp_b2   = (const float*)d_in[15];
  const float* lnf_g    = (const float*)d_in[16];
  const float* lnf_b    = (const float*)d_in[17];
  float* out = (float*)d_out;

  // Workspace carve (256B aligned)
  uintptr_t w = (uintptr_t)d_ws;
  auto carve = [&](size_t bytes) -> void* {
    void* p = (void*)w;
    w += (bytes + 255) & ~(size_t)255;
    return p;
  };
  const size_t szAttn = (size_t)Lc * 3 * Cc * Cc;   // 2304x768 per layer, N x K
  const size_t szProj = (size_t)Lc * Cc * Cc;
  const size_t szM1   = (size_t)Lc * 4 * Cc * Cc;   // 3072x768
  const size_t szM2   = (size_t)Lc * 4 * Cc * Cc;   // 768x3072
  bf16_t* wT_attn = (bf16_t*)carve(szAttn * 2);
  bf16_t* wT_proj = (bf16_t*)carve(szProj * 2);
  bf16_t* wT_m1   = (bf16_t*)carve(szM1 * 2);
  bf16_t* wT_m2   = (bf16_t*)carve(szM2 * 2);
  float*  x    = (float*) carve((size_t)Bc * TMAX * Cc * 4);
  bf16_t* hbf  = (bf16_t*)carve((size_t)Bc * TMAX * Cc * 2);
  float*  qkv  = (float*) carve((size_t)Bc * TMAX * 3 * Cc * 4);
  bf16_t* obf  = (bf16_t*)carve((size_t)Bc * TMAX * Cc * 2);
  bf16_t* ubf  = (bf16_t*)carve((size_t)Bc * TMAX * 4 * Cc * 2);
  float*  cot  = (float*) carve((size_t)NCOT * Bc * Cc * 4);

  // One-time fp32 -> bf16 transposed weight conversion (L2-resident thereafter)
  {
    auto tgrid = [](size_t total) { return (unsigned)((total + 255) / 256); };
    transpose_to_bf16<<<tgrid(szAttn), 256, 0, stream>>>(attn_w, wT_attn, Cc, 3 * Cc, Lc);
    transpose_to_bf16<<<tgrid(szProj), 256, 0, stream>>>(proj_w, wT_proj, Cc, Cc, Lc);
    transpose_to_bf16<<<tgrid(szM1),   256, 0, stream>>>(mlp_w1, wT_m1, Cc, 4 * Cc, Lc);
    transpose_to_bf16<<<tgrid(szM2),   256, 0, stream>>>(mlp_w2, wT_m2, 4 * Cc, Cc, Lc);
  }

  for (int p = 0; p <= NCOT; ++p) {
    const int T = 81 + p + 1;       // 82..86
    const int Mr = Bc * T;          // rows, always multiple of 16
    const dim3 gB((unsigned)((Mr + 127) / 128), 1, 1);

    embed_kernel<<<(unsigned)((Bc * T * Cc + 255) / 256), 256, 0, stream>>>(
        idx, wte, special, cot, T, x);

    for (int l = 0; l < Lc; ++l) {
      layernorm_bf16<<<(unsigned)Mr, 256, 0, stream>>>(
          x, ln1_g + (size_t)l * Cc, ln1_b + (size_t)l * Cc, hbf);

      gemm_bf16_wmma<0><<<dim3(gB.x, 3 * Cc / 64), 256, 0, stream>>>(
          hbf, wT_attn + (size_t)l * 3 * Cc * Cc, attn_b + (size_t)l * 3 * Cc,
          nullptr, qkv, nullptr, Mr, Cc, 3 * Cc);

      attention_kernel<<<(unsigned)(Bc * 12), 128, 0, stream>>>(qkv, obf, T);

      gemm_bf16_wmma<1><<<dim3(gB.x, Cc / 64), 256, 0, stream>>>(
          obf, wT_proj + (size_t)l * Cc * Cc, proj_b + (size_t)l * Cc,
          x, x, nullptr, Mr, Cc, Cc);

      layernorm_bf16<<<(unsigned)Mr, 256, 0, stream>>>(
          x, ln2_g + (size_t)l * Cc, ln2_b + (size_t)l * Cc, hbf);

      gemm_bf16_wmma<2><<<dim3(gB.x, 4 * Cc / 64), 256, 0, stream>>>(
          hbf, wT_m1 + (size_t)l * 4 * Cc * Cc, mlp_b1 + (size_t)l * 4 * Cc,
          nullptr, nullptr, ubf, Mr, Cc, 4 * Cc);

      gemm_bf16_wmma<1><<<dim3(gB.x, Cc / 64), 256, 0, stream>>>(
          ubf, wT_m2 + (size_t)l * 4 * Cc * Cc, mlp_b2 + (size_t)l * Cc,
          x, x, nullptr, Mr, 4 * Cc, Cc);
    }

    float* dst = (p < NCOT) ? (cot + (size_t)p * Bc * Cc) : out;
    lnf_last<<<(unsigned)Bc, 256, 0, stream>>>(x, lnf_g, lnf_b, T, dst);
  }
}
